// MambaCompressor_7370163880481
// MI455X (gfx1250) — matrix operand order repaired and checked by reading, not compiled, not run on hardware
//
#include <hip/hip_runtime.h>

// ---------------- problem constants ----------------
static constexpr int Bn   = 4;
static constexpr int Ln   = 1024;
static constexpr int Hn   = 1024;
static constexpr int DIn  = 2048;
static constexpr int DSn  = 16;
static constexpr int DTRn = 64;
static constexpr int NLn  = 4;
static constexpr int LLMn = 4096;
static constexpr int Kn   = 16;       // memory tokens per row (L/STRIDE)
static constexpr int STRIDEn = 64;
static constexpr float EPSn = 1e-5f;

typedef __attribute__((ext_vector_type(16))) __bf16 v16bf;
typedef __attribute__((ext_vector_type(8)))  __bf16 v8bf;
typedef __attribute__((ext_vector_type(8)))  float  v8f;

__device__ __forceinline__ __bf16 to_bf16(float f) { return (__bf16)f; }
__device__ __forceinline__ float silu(float v) { return v / (1.f + __expf(-v)); }

// ---------------- f32 -> bf16 weight conversion ----------------
__global__ void f32_to_bf16_kernel(const float* __restrict__ in, __bf16* __restrict__ out, int n) {
    int i = blockIdx.x * blockDim.x + threadIdx.x;
    if (i < n) out[i] = to_bf16(in[i]);
}

// ---------------- embedding gather ----------------
__global__ void embed_kernel(const int* __restrict__ ids, const float* __restrict__ embed,
                             float* __restrict__ h) {
    int i = blockIdx.x * blockDim.x + threadIdx.x;       // over B*L*H
    int row = i / Hn;                                    // b*L + t
    int col = i - row * Hn;
    int tok = ids[row];
    h[i] = embed[(size_t)tok * Hn + col];
}

// ---------------- rmsnorm -> bf16 (one block per row) ----------------
__global__ void rmsnorm_bf16_kernel(const float* __restrict__ h, const float* __restrict__ w,
                                    __bf16* __restrict__ out) {
    __shared__ float red[256];
    const size_t base = (size_t)blockIdx.x * Hn;
    float ss = 0.f;
    for (int i = threadIdx.x; i < Hn; i += blockDim.x) { float v = h[base + i]; ss += v * v; }
    red[threadIdx.x] = ss; __syncthreads();
    for (int s = 128; s > 0; s >>= 1) {
        if ((int)threadIdx.x < s) red[threadIdx.x] += red[threadIdx.x + s];
        __syncthreads();
    }
    const float scale = rsqrtf(red[0] / (float)Hn + EPSn);
    for (int i = threadIdx.x; i < Hn; i += blockDim.x)
        out[base + i] = to_bf16(h[base + i] * scale * w[i]);
}

// ---------------- final rmsnorm + gather memory tokens ----------------
__global__ void gather_rmsnorm_kernel(const float* __restrict__ h, const float* __restrict__ w,
                                      __bf16* __restrict__ feats) {
    // blockIdx.x in [0, B*Kn): row b, memory token k -> t = STRIDE-1 + k*STRIDE
    __shared__ float red[256];
    const int b = blockIdx.x / Kn;
    const int k = blockIdx.x - b * Kn;
    const int t = STRIDEn - 1 + k * STRIDEn;
    const size_t src = ((size_t)b * Ln + t) * Hn;
    const size_t dst = (size_t)blockIdx.x * Hn;
    float ss = 0.f;
    for (int i = threadIdx.x; i < Hn; i += blockDim.x) { float v = h[src + i]; ss += v * v; }
    red[threadIdx.x] = ss; __syncthreads();
    for (int s = 128; s > 0; s >>= 1) {
        if ((int)threadIdx.x < s) red[threadIdx.x] += red[threadIdx.x + s];
        __syncthreads();
    }
    const float scale = rsqrtf(red[0] / (float)Hn + EPSn);
    for (int i = threadIdx.x; i < Hn; i += blockDim.x)
        feats[dst + i] = to_bf16(h[src + i] * scale * w[i]);
}

// ---------------- WMMA bf16 GEMM: C[M,N] = A[M,K] * W[N,K]^T (+res)(+bias) ----
// One wave owns a (16*MT) x (16*NT) output tile: MT x NT f32 accumulators.
// Each k-step: MT A-fragment loads + NT B-fragment loads feed MT*NT WMMAs,
// raising L2 arithmetic intensity ~(MT*NT)/(MT+NT) and giving deep
// independent-WMMA ILP. __launch_bounds__(256, 2) relaxes the occupancy
// target to 2 waves/SIMD so the 16 accumulators (128 VGPRs) + fragments stay
// register-resident instead of spilling to scratch.
template <int MT, int NT>
__global__ __launch_bounds__(256, 2)
void gemm_bf16_kernel(const __bf16* __restrict__ A,   // M x K row-major (bf16)
                      const __bf16* __restrict__ W,   // N x K row-major (bf16)
                      float* C,                       // M x N (f32)
                      const float* residual,          // M x N or nullptr
                      const float* __restrict__ bias, // N or nullptr
                      int M, int N, int K) {
    const int lane  = threadIdx.x & 31;
    const int wave  = threadIdx.x >> 5;
    const int waves = blockDim.x >> 5;
    const int tile  = blockIdx.x * waves + wave;
    const int nTilesN = N / (16 * NT);
    const int m0 = (tile / nTilesN) * (16 * MT);
    const int n0 = (tile % nTilesN) * (16 * NT);
    if (m0 >= M) return;

    const int row  = lane & 15;          // A: M row, W: N column, D: N column
    const int aoff = (lane >> 4) * 16;   // A K-half select
    const int boff = (lane >> 4) * 16;   // B K-half select

    v8f acc[MT][NT];
#pragma unroll
    for (int mt = 0; mt < MT; ++mt)
#pragma unroll
        for (int nt = 0; nt < NT; ++nt)
#pragma unroll
            for (int r = 0; r < 8; ++r) acc[mt][nt][r] = 0.f;

    const __bf16* Abase = A + (size_t)(m0 + row) * K;
    for (int k = 0; k < K; k += 32) {
        v16bf a[MT];
#pragma unroll
        for (int mt = 0; mt < MT; ++mt) {
            const __bf16* Arow = Abase + (size_t)(mt * 16) * K + k;
            const v8bf a0 = *reinterpret_cast<const v8bf*>(Arow + aoff);
            const v8bf a1 = *reinterpret_cast<const v8bf*>(Arow + 8 + aoff);
#pragma unroll
            for (int i = 0; i < 8; ++i) { a[mt][i] = a0[i]; a[mt][8 + i] = a1[i]; }
        }
#pragma unroll
        for (int nt = 0; nt < NT; ++nt) {
            const __bf16* Wrow = W + (size_t)(n0 + nt * 16 + row) * K + k + boff;
            const v16bf b = *reinterpret_cast<const v16bf*>(Wrow);
#pragma unroll
            for (int mt = 0; mt < MT; ++mt)
                acc[mt][nt] = __builtin_amdgcn_wmma_f32_16x16x32_bf16(
                    false, a[mt], false, b, (short)0, acc[mt][nt], false, false);
        }
    }

    const int nCol = lane & 15;
#pragma unroll
    for (int mt = 0; mt < MT; ++mt) {
        const int mBase = m0 + mt * 16 + (lane >> 4) * 8;
#pragma unroll
        for (int nt = 0; nt < NT; ++nt) {
            const int nn = n0 + nt * 16 + nCol;
            const float bv = bias ? bias[nn] : 0.f;
#pragma unroll
            for (int r = 0; r < 8; ++r) {
                const size_t idx = (size_t)(mBase + r) * N + nn;
                float v = acc[mt][nt][r] + bv;
                if (residual) v += residual[idx];
                C[idx] = v;
            }
        }
    }
}

// ---------------- depthwise causal conv(4) + bias + silu ----------------
__global__ void conv_silu_kernel(const float* __restrict__ xz,  // (B,L,2*DI); x = [:DI]
                                 const float* __restrict__ cw,  // (DI,4)
                                 const float* __restrict__ cb,  // (DI)
                                 float* __restrict__ u,         // (B,L,DI) f32
                                 __bf16* __restrict__ u_bf) {   // (B,L,DI) bf16
    int i = blockIdx.x * blockDim.x + threadIdx.x;   // over B*L*DI
    int d   = i % DIn;
    int bt  = i / DIn;
    int t   = bt % Ln;
    int b   = bt / Ln;
    float acc = cb[d];
#pragma unroll
    for (int j = 0; j < 4; ++j) {
        int ts = t - 3 + j;
        if (ts >= 0) acc += xz[((size_t)(b * Ln + ts)) * (2 * DIn) + d] * cw[d * 4 + j];
    }
    float v = silu(acc);
    u[i] = v;
    u_bf[i] = to_bf16(v);
}

// ---------------- split dt_raw out of ssm, convert to bf16 ----------------
__global__ void split_dt_kernel(const float* __restrict__ ssm, __bf16* __restrict__ dtraw_bf) {
    int i = blockIdx.x * blockDim.x + threadIdx.x;   // over B*L*DTR
    int r = i / DTRn;
    int j = i - r * DTRn;
    dtraw_bf[i] = to_bf16(ssm[(size_t)r * (DTRn + 2 * DSn) + j]);
}

// ---------------- selective scan (one thread per (b,d) channel) ----------------
static constexpr int SCAN_CHUNK = 64;
__global__ void scan_kernel(const float* __restrict__ ssm,    // (B,L,96): [64:80)=Bm [80:96)=Cm
                            const float* __restrict__ dtlin,  // (B,L,DI) pre-bias dt
                            const float* __restrict__ dtb,    // (DI)
                            const float* __restrict__ Alog,   // (DI,DS)
                            const float* __restrict__ Dp,     // (DI)
                            const float* __restrict__ u,      // (B,L,DI)
                            const float* __restrict__ xz,     // (B,L,2*DI): z = [DI:]
                            __bf16* __restrict__ y_bf) {      // (B,L,DI)
    const int b = blockIdx.y;
    const int d = blockIdx.x * blockDim.x + threadIdx.x;

    float A[DSn], st[DSn];
#pragma unroll
    for (int s = 0; s < DSn; ++s) { A[s] = -__expf(Alog[d * DSn + s]); st[s] = 0.f; }
    const float db = dtb[d];
    const float Dd = Dp[d];

    __shared__ float sB[SCAN_CHUNK][DSn];
    __shared__ float sC[SCAN_CHUNK][DSn];

    for (int t0 = 0; t0 < Ln; t0 += SCAN_CHUNK) {
        __syncthreads();
        for (int i = threadIdx.x; i < SCAN_CHUNK * DSn; i += blockDim.x) {
            int tt = i / DSn, s = i - tt * DSn;
            const size_t rb = ((size_t)(b * Ln + t0 + tt)) * 96;
            sB[tt][s] = ssm[rb + 64 + s];
            sC[tt][s] = ssm[rb + 80 + s];
        }
        __syncthreads();
        for (int tt = 0; tt < SCAN_CHUNK; ++tt) {
            const int t = t0 + tt;
            const size_t ix = ((size_t)(b * Ln + t)) * DIn + d;
            float dtv = dtlin[ix] + db;
            dtv = (dtv > 20.f) ? dtv : log1pf(__expf(dtv));       // softplus
            const float uv = u[ix];
            const float du = dtv * uv;
            float y = 0.f;
#pragma unroll
            for (int s = 0; s < DSn; ++s) {
                const float dA = __expf(dtv * A[s]);
                st[s] = st[s] * dA + du * sB[tt][s];
                y += st[s] * sC[tt][s];
            }
            y += uv * Dd;
            const float zv = xz[((size_t)(b * Ln + t)) * (2 * DIn) + DIn + d];
            y *= silu(zv);
            y_bf[ix] = to_bf16(y);
        }
    }
}

// ---------------- host side ----------------
static inline size_t alignUp(size_t x) { return (x + 255) & ~(size_t)255; }
static inline int gemmBlocks(int M, int N, int MT, int NT) {
    return (M / (16 * MT)) * (N / (16 * NT)) / 8;
}

extern "C" void kernel_launch(void* const* d_in, const int* in_sizes, int n_in,
                              void* d_out, int out_size, void* d_ws, size_t ws_size,
                              hipStream_t stream) {
    (void)in_sizes; (void)n_in; (void)out_size; (void)ws_size;
    const int*   ids     = (const int*)  d_in[0];
    const float* embed   = (const float*)d_in[1];
    const float* norm_w  = (const float*)d_in[2];
    const float* in_w    = (const float*)d_in[3];   // (NL, 2*DI, H)
    const float* conv_w  = (const float*)d_in[4];   // (NL, DI, 4)
    const float* conv_b  = (const float*)d_in[5];   // (NL, DI)
    const float* x_w     = (const float*)d_in[6];   // (NL, 96, DI)
    const float* dt_w    = (const float*)d_in[7];   // (NL, DI, DTR)
    const float* dt_b    = (const float*)d_in[8];   // (NL, DI)
    const float* A_log   = (const float*)d_in[9];   // (NL, DI, DS)
    const float* Dp      = (const float*)d_in[10];  // (NL, DI)
    const float* out_w   = (const float*)d_in[11];  // (NL, H, DI)
    const float* fnorm_w = (const float*)d_in[12];  // (H)
    const float* proj_w  = (const float*)d_in[13];  // (LLM, H)
    const float* proj_b  = (const float*)d_in[14];  // (LLM)
    float* out = (float*)d_out;

    // workspace carve-up
    char* p = (char*)d_ws;
    auto take = [&](size_t bytes) { char* r = p; p += alignUp(bytes); return r; };
    const int ML = Bn * Ln;                                   // 4096 rows
    float*  h        = (float*) take((size_t)ML * Hn * 4);
    float*  xz       = (float*) take((size_t)ML * 2 * DIn * 4);
    __bf16* x_bf     = (__bf16*)take((size_t)ML * Hn * 2);
    float*  u        = (float*) take((size_t)ML * DIn * 4);
    __bf16* u_bf     = (__bf16*)take((size_t)ML * DIn * 2);
    float*  ssm      = (float*) take((size_t)ML * 96 * 4);
    __bf16* dtraw_bf = (__bf16*)take((size_t)ML * DTRn * 2);
    float*  dtlin    = (float*) take((size_t)ML * DIn * 4);
    __bf16* y_bf     = (__bf16*)take((size_t)ML * DIn * 2);
    __bf16* feats_bf = (__bf16*)take((size_t)Bn * Kn * Hn * 2);
    __bf16* inw_bf   = (__bf16*)take((size_t)NLn * 2 * DIn * Hn * 2);
    __bf16* xw_bf    = (__bf16*)take((size_t)NLn * 96 * DIn * 2);
    __bf16* dtw_bf   = (__bf16*)take((size_t)NLn * DIn * DTRn * 2);
    __bf16* outw_bf  = (__bf16*)take((size_t)NLn * Hn * DIn * 2);
    __bf16* projw_bf = (__bf16*)take((size_t)LLMn * Hn * 2);

    const int T = 256;
    auto cvt = [&](const float* s, __bf16* d, int n) {
        f32_to_bf16_kernel<<<(n + T - 1) / T, T, 0, stream>>>(s, d, n);
    };
    cvt(in_w,   inw_bf,   NLn * 2 * DIn * Hn);
    cvt(x_w,    xw_bf,    NLn * 96 * DIn);
    cvt(dt_w,   dtw_bf,   NLn * DIn * DTRn);
    cvt(out_w,  outw_bf,  NLn * Hn * DIn);
    cvt(proj_w, projw_bf, LLMn * Hn);

    embed_kernel<<<(ML * Hn) / T, T, 0, stream>>>(ids, embed, h);

    for (int l = 0; l < NLn; ++l) {
        // x = rmsnorm(h) -> bf16
        rmsnorm_bf16_kernel<<<ML, T, 0, stream>>>(h, norm_w + (size_t)l * Hn, x_bf);
        // xz = x @ in_w^T                          (4096 x 4096 x 1024), 64x64 wave tiles
        gemm_bf16_kernel<4, 4><<<gemmBlocks(ML, 2 * DIn, 4, 4), T, 0, stream>>>(
            x_bf, inw_bf + (size_t)l * 2 * DIn * Hn, xz, nullptr, nullptr, ML, 2 * DIn, Hn);
        // u = silu(conv(x) + cb)
        conv_silu_kernel<<<(ML * DIn) / T, T, 0, stream>>>(
            xz, conv_w + (size_t)l * DIn * 4, conv_b + (size_t)l * DIn, u, u_bf);
        // ssm = u @ x_w^T                          (4096 x 96 x 2048), tall-skinny
        gemm_bf16_kernel<1, 2><<<gemmBlocks(ML, 96, 1, 2), T, 0, stream>>>(
            u_bf, xw_bf + (size_t)l * 96 * DIn, ssm, nullptr, nullptr, ML, 96, DIn);
        // dt_raw -> bf16
        split_dt_kernel<<<(ML * DTRn) / T, T, 0, stream>>>(ssm, dtraw_bf);
        // dtlin = dt_raw @ dt_w^T                  (4096 x 2048 x 64), 64x64 wave tiles
        gemm_bf16_kernel<4, 4><<<gemmBlocks(ML, DIn, 4, 4), T, 0, stream>>>(
            dtraw_bf, dtw_bf + (size_t)l * DIn * DTRn, dtlin, nullptr, nullptr, ML, DIn, DTRn);
        // selective scan + gating -> y_bf
        scan_kernel<<<dim3(DIn / T, Bn), T, 0, stream>>>(
            ssm, dtlin, dt_b + (size_t)l * DIn, A_log + (size_t)l * DIn * DSn,
            Dp + (size_t)l * DIn, u, xz, y_bf);
        // h = h + y @ out_w^T                      (4096 x 1024 x 2048), 64x64 wave tiles
        gemm_bf16_kernel<4, 4><<<gemmBlocks(ML, Hn, 4, 4), T, 0, stream>>>(
            y_bf, outw_bf + (size_t)l * Hn * DIn, h, h, nullptr, ML, Hn, DIn);
    }

    // final rmsnorm at the 16 memory-token positions per batch row
    gather_rmsnorm_kernel<<<Bn * Kn, T, 0, stream>>>(h, fnorm_w, feats_bf);
    // out = feats @ proj_w^T + proj_b              (64 x 4096 x 1024), wide-N small-M
    gemm_bf16_kernel<1, 4><<<gemmBlocks(Bn * Kn, LLMn, 1, 4), T, 0, stream>>>(
        feats_bf, projw_bf, out, nullptr, proj_b, Bn * Kn, LLMn, Hn);
}
